// ConvCapsuleLayer3D_37177236914807
// MI455X (gfx1250) — matrix operand order repaired and checked by reading, not compiled, not run on hardware
//
#include <hip/hip_runtime.h>
#include <hip/hip_bf16.h>

typedef __attribute__((ext_vector_type(16))) _Float16 v16h;
typedef __attribute__((ext_vector_type(8)))  _Float16 v8h;
typedef __attribute__((ext_vector_type(8)))  float    v8f;

// ---------------- constants from the reference shapes ----------------
#define Bsz   32
#define IC    8
#define IA    16
#define OC    16
#define NA    16
#define NN    256          // OC*NA
#define HC    30
#define WC    30
#define PIX   900          // HC*WC
#define KRED  144          // IA*3*3
#define KP    160          // K padded to 5 chunks of 32
#define KS    168          // LDS K stride (keeps 16B alignment: 168*2=336=21*16)
#define MT    64           // pixels per block in kernel 1
#define NPT   15           // ceil(900/64)

// ---------------- kernel 0: W [144][256] fp32 -> Wh [256][160] fp16 (transposed, K zero-padded)
__global__ void capsW_to_fp16(const float* __restrict__ W, _Float16* __restrict__ Wh) {
    int idx = blockIdx.x * 256 + threadIdx.x;
    if (idx < NN * KP) {
        int n = idx / KP, kp = idx - n * KP;
        float v = (kp < KRED) ? W[kp * NN + n] : 0.0f;
        Wh[idx] = (_Float16)v;
    }
}

// ---------------- kernel 1: votes GEMM via WMMA ----------------------
// votes layout: votes[b][i][p][n]  (p = y*30+x, n = o*16+a), fp16
__global__ void caps_votes_wmma(const float* __restrict__ xg,
                                const _Float16* __restrict__ Wh,
                                _Float16* __restrict__ votes) {
    extern __shared__ char smem[];
    _Float16* As = (_Float16*)smem;               // [MT][KS] fp16 im2col patch

    const int ptile = blockIdx.x;                 // 0..14
    const int i     = blockIdx.y;                 // input capsule
    const int b     = blockIdx.z;                 // batch
    const unsigned tid = threadIdx.x;             // 256 threads = 8 waves

    // ---- im2col into LDS (fp16), zero-padded in K and for p>=900 ----
    const size_t xbase = (size_t)b * (IC * IA * 32 * 32);
    for (int idx = tid; idx < MT * KP; idx += 256) {
        int m = idx / KP, k = idx - m * KP;
        int p = ptile * MT + m;
        float val = 0.0f;
        if (k < KRED && p < PIX) {
            int da = k / 9;
            int r  = k - da * 9;
            int ky = r / 3, kx = r - (r / 3) * 3;
            int y  = p / WC, x = p - (p / WC) * WC;
            val = xg[xbase + (size_t)(i * IA + da) * 1024 + (y + ky) * 32 + (x + kx)];
        }
        As[m * KS + k] = (_Float16)val;
    }
    __syncthreads();

    const unsigned lane  = tid & 31u;
    const unsigned wave  = tid >> 5;              // 0..7
    const unsigned mtile = wave & 3u;             // 4 M-tiles of 16 rows
    const unsigned nhalf = wave >> 2;             // N 0..127 / 128..255

    v8f acc[8] = {};

    #pragma unroll
    for (int kc = 0; kc < 5; ++kc) {
        // A fragment (16x32 f16): lane L row = L%16; VGPR0-3 hold K = 8*(L/16)+0..7,
        // VGPR4-7 hold K = 16 + 8*(L/16)+0..7  (ISA 7.12.2)
        int kg = kc * 32 + ((lane >> 4) << 3);
        const _Float16* ap = &As[(mtile * 16 + (lane & 15u)) * KS + kg];
        v8h alo = *(const v8h*)ap;
        v8h ahi = *(const v8h*)(ap + 16);
        v16h a = __builtin_shufflevector(alo, ahi,
                 0,1,2,3,4,5,6,7,8,9,10,11,12,13,14,15);

        // B fragment (32x16 f16): lane L col = L%16; K = 8*(L/16)+0..7 and
        // 16 + 8*(L/16)+0..7 -> two 16B loads from Wh[n][k] layout
        int kb = kc * 32 + ((lane >> 4) << 3);
        #pragma unroll
        for (int nt = 0; nt < 8; ++nt) {
            int n = (int)(nhalf * 8 + nt) * 16 + (int)(lane & 15u);
            v8h blo = *(const v8h*)(Wh + n * KP + kb);
            v8h bhi = *(const v8h*)(Wh + n * KP + kb + 16);
            v16h bf = __builtin_shufflevector(blo, bhi,
                      0,1,2,3,4,5,6,7,8,9,10,11,12,13,14,15);
            acc[nt] = __builtin_amdgcn_wmma_f32_16x16x32_f16(
                          false, a, false, bf, (short)0, acc[nt], false, false);
        }
    }

    // ---- store votes as fp16: D 16x16 layout: lane N=L%16, VGPR v M=v+8*(L/16)
    const size_t vbase = (size_t)(b * IC + i) * PIX;
    #pragma unroll
    for (int nt = 0; nt < 8; ++nt) {
        int n = (int)(nhalf * 8 + nt) * 16 + (int)(lane & 15u);
        #pragma unroll
        for (int v = 0; v < 8; ++v) {
            int mloc = (int)(mtile * 16) + v + (int)((lane >> 4) << 3);
            int p = ptile * MT + mloc;
            if (p < PIX)
                votes[(vbase + p) * NN + n] = (_Float16)acc[nt][v];
        }
    }
}

// ---------------- kernel 2: dynamic routing entirely in LDS ----------
// block = (y, b); LDS: vs[8][30*256] fp16, logits[3840], route[3840], pre[7680] f32
__global__ void caps_routing(const _Float16* __restrict__ votes,
                             const float* __restrict__ bias,   // [16][16]
                             float* __restrict__ out) {
    extern __shared__ char smem[];
    _Float16* vs  = (_Float16*)smem;                          // 61440 fp16
    float* logits = (float*)(smem + IC * WC * NN * 2);        // 3840 f32
    float* route  = logits + IC * OC * WC;                    // 3840 f32
    float* pre    = route  + IC * OC * WC;                    // 7680 f32

    const int y = blockIdx.x, b = blockIdx.y;
    const unsigned tid = threadIdx.x;

    // ---- load votes slice [i][x][oa] once (16B vector copies) ----
    for (int t = tid; t < IC * (WC * NN / 8); t += 256) {      // 8 f16 per uint4
        int i = t / (WC * NN / 8), j = t - i * (WC * NN / 8);
        ((uint4*)vs)[i * (WC * NN / 8) + j] =
            *(const uint4*)(votes + ((size_t)(b * IC + i) * PIX + y * WC) * NN + j * 8);
    }
    for (int t = tid; t < IC * OC * WC; t += 256) logits[t] = 0.0f;
    __syncthreads();

    for (int r = 0; r < 3; ++r) {
        // softmax over output capsules o, per (i, x)
        for (int it = tid; it < IC * WC; it += 256) {
            int i = it / WC, x = it - (it / WC) * WC;
            float l[OC], mx = -1e30f;
            #pragma unroll
            for (int o = 0; o < OC; ++o) { l[o] = logits[(i * OC + o) * WC + x]; mx = fmaxf(mx, l[o]); }
            float s = 0.0f;
            #pragma unroll
            for (int o = 0; o < OC; ++o) { l[o] = __expf(l[o] - mx); s += l[o]; }
            float inv = 1.0f / s;
            #pragma unroll
            for (int o = 0; o < OC; ++o) route[(i * OC + o) * WC + x] = l[o] * inv;
        }
        __syncthreads();

        // preact[e] = bias + sum_i route[i][o][x] * vs[i][e],  e = x*256 + o*16 + a
        for (int e = tid; e < WC * NN; e += 256) {
            int x = e >> 8, oa = e & 255, o = oa >> 4;
            float acc = bias[oa];
            #pragma unroll
            for (int i = 0; i < IC; ++i)
                acc += route[(i * OC + o) * WC + x] * (float)vs[i * (WC * NN) + e];
            pre[e] = acc;
        }
        __syncthreads();

        // squash over spatial-x axis (reference axis=-1), per (o,a); in-place
        {
            int oa = tid;                                     // exactly 256 threads
            float s2 = 0.0f;
            for (int x = 0; x < WC; ++x) { float v = pre[x * NN + oa]; s2 += v * v; }
            float scale = (s2 / (1.0f + s2)) * rsqrtf(s2 + 1e-7f);
            for (int x = 0; x < WC; ++x) pre[x * NN + oa] *= scale;
        }
        __syncthreads();

        if (r < 2) {
            // logits[i][o][x] += sum_a vs[i][x][o][a] * act[x][o][a]
            for (int it = tid; it < IC * OC * WC; it += 256) {
                int x = it % WC, t2 = it / WC, o = t2 & 15, i = t2 >> 4;
                const _Float16* vp = vs + i * (WC * NN) + x * NN + o * NA;
                const float*    pp = pre + x * NN + o * NA;
                float s = 0.0f;
                #pragma unroll
                for (int a = 0; a < NA; ++a) s += (float)vp[a] * pp[a];
                logits[it] += s;
            }
            __syncthreads();
        }
    }

    // out[b][y][x][o][a] contiguous == pre[e]
    float* outp = out + (size_t)((b * HC + y) * WC) * NN;
    for (int e = tid; e < WC * NN; e += 256) outp[e] = pre[e];
}

// ---------------- host launcher --------------------------------------
extern "C" void kernel_launch(void* const* d_in, const int* in_sizes, int n_in,
                              void* d_out, int out_size, void* d_ws, size_t ws_size,
                              hipStream_t stream) {
    const float* x    = (const float*)d_in[0];
    const float* W    = (const float*)d_in[1];
    const float* bias = (const float*)d_in[2];
    float* out = (float*)d_out;

    _Float16* Wh    = (_Float16*)d_ws;            // 40,960 fp16 = 80 KB
    _Float16* votes = Wh + NN * KP;               // 58,982,400 fp16 ~= 118 MB

    (void)in_sizes; (void)n_in; (void)out_size; (void)ws_size;

    size_t smem1 = (size_t)MT * KS * sizeof(_Float16);                  // 21,504 B
    size_t smem2 = (size_t)IC * WC * NN * 2 + (2 * IC * OC * WC + WC * NN) * 4; // 184,320 B
    hipFuncSetAttribute((const void*)caps_routing,
                        hipFuncAttributeMaxDynamicSharedMemorySize, (int)smem2);

    capsW_to_fp16<<<(NN * KP + 255) / 256, 256, 0, stream>>>(W, Wh);
    caps_votes_wmma<<<dim3(NPT, IC, Bsz), 256, smem1, stream>>>(x, Wh, votes);
    caps_routing<<<dim3(HC, Bsz), 256, smem2, stream>>>(votes, bias, out);
}
